// HGTRegressor_88605175316550
// MI455X (gfx1250) — matrix operand order repaired and checked by reading, not compiled, not run on hardware
//
#include <hip/hip_runtime.h>
#include <hip/hip_bf16.h>

typedef __attribute__((ext_vector_type(16))) __bf16 v16bf;
typedef __attribute__((ext_vector_type(8)))  float  v8f;
typedef __attribute__((ext_vector_type(4)))  unsigned int v4u;

#define NGENE 50000
#define NSNP  50000
#define NE    250000

// ---------------- helpers ----------------
__device__ __forceinline__ unsigned short f2bf(float f) {
  unsigned int u = __float_as_uint(f);
  unsigned int r = u + 0x7FFFu + ((u >> 16) & 1u);   // round-to-nearest-even
  return (unsigned short)(r >> 16);
}
__device__ __forceinline__ float geluf(float x) {
  return 0.5f * x * (1.0f + erff(x * 0.70710678118654752f));
}
__device__ __forceinline__ void atomicMaxF(float* a, float v) {
  unsigned int* ua = (unsigned int*)a;
  unsigned int old = *ua;
  while (__uint_as_float(old) < v) {
    unsigned int assumed = old;
    old = atomicCAS(ua, assumed, __float_as_uint(v));
    if (old == assumed) break;
  }
}

// ---------------- WMMA GEMM: C = epi(act_in(A) @ B + bias) ----------------
// A: [M,K] f32 (lda), B: [K,N] f32 (ldb) or [N,K] if transB, C: [M,N] f32 (ldc)
// in_act: 0 none, 2 gelu (applied to A elements).  out_act: 0 none, 1 relu.
// beta: 1 -> C += result.  gate/skip: C = sig(gate)*res + (1-sig)*skip.
#define BM 64
#define BN 64
#define BK 32
#define AST 40   // LDS stride in bf16: 20 dwords -> 16B-aligned frags, conflict-free
#define BST 40

__launch_bounds__(128)
__global__ void gemm_wmma(const float* __restrict__ A, int lda,
                          const float* __restrict__ B, int ldb, int transB,
                          const float* __restrict__ bias,
                          const float* __restrict__ skip,
                          const float* __restrict__ gate,
                          float* __restrict__ C, int ldc,
                          int M, int N, int K,
                          int in_act, int out_act, int beta) {
  __shared__ __attribute__((aligned(16))) unsigned int AsU[(BM * AST) / 2];
  __shared__ __attribute__((aligned(16))) unsigned int BsU[(BN * BST) / 2];
  unsigned short* Ash = (unsigned short*)AsU;
  unsigned short* Bsh = (unsigned short*)BsU;

  const int tid  = threadIdx.x;
  const int wv   = tid >> 5;          // wave id 0..3 -> 16-row slice
  const int lane = tid & 31;
  const int hlf  = lane >> 4;
  const int mr   = lane & 15;
  const int rowBase = blockIdx.y * BM;
  const int colBase = blockIdx.x * BN;

  v8f acc[4];
#pragma unroll
  for (int i = 0; i < 4; ++i)
#pragma unroll
    for (int j = 0; j < 8; ++j) acc[i][j] = 0.0f;

  union U { v4u q[2]; v16bf v; };
  // fragment LDS dword bases (contiguous k = 16*hlf + s in both A and B, b128-friendly)
  const int aBase = (wv * 16 + mr) * (AST / 2) + 8 * hlf;   // + 4 for next 8 K

  for (int k0 = 0; k0 < K; k0 += BK) {
    __syncthreads();
    // stage A tile (64x32) as bf16, fused input activation
#pragma unroll
    for (int i = 0; i < 16; ++i) {
      int idx = tid + i * 128;
      int r = idx >> 5, c = idx & 31;
      int gr = rowBase + r;
      float v = (gr < M) ? A[(size_t)gr * lda + (k0 + c)] : 0.0f;
      if (in_act == 2) v = geluf(v);
      Ash[r * AST + c] = f2bf(v);
    }
    // stage B tile as Bs[n][k] (N-major so K runs contiguously per column)
    if (transB) {
#pragma unroll
      for (int i = 0; i < 16; ++i) {
        int idx = tid + i * 128;
        int k = idx & 31, n = idx >> 5;
        Bsh[n * BST + k] = f2bf(B[(size_t)(colBase + n) * ldb + (k0 + k)]);
      }
    } else {
#pragma unroll
      for (int i = 0; i < 16; ++i) {
        int idx = tid + i * 128;
        int n = idx & 63, k = idx >> 6;
        Bsh[n * BST + k] = f2bf(B[(size_t)(k0 + k) * ldb + (colBase + n)]);
      }
    }
    // prefetch next tiles into cache while this step computes
    if (k0 + BK < K) {
      int gr = rowBase + (tid >> 5);
      if (gr < M) __builtin_prefetch(&A[(size_t)gr * lda + (k0 + BK)], 0, 3);
      if (transB)
        __builtin_prefetch(&B[(size_t)(colBase + (tid >> 1)) * ldb + (k0 + BK)], 0, 3);
      else
        __builtin_prefetch(&B[(size_t)(k0 + BK + (tid >> 2)) * ldb + colBase], 0, 3);
    }
    __syncthreads();

    // load all fragments first (b128 LDS loads), then stream 4 WMMAs
    U bf0, bf1, bf2, bf3, af;
    {
      const int b0 = (0 * 16 + mr) * (BST / 2) + 8 * hlf;
      const int b1 = (1 * 16 + mr) * (BST / 2) + 8 * hlf;
      const int b2 = (2 * 16 + mr) * (BST / 2) + 8 * hlf;
      const int b3 = (3 * 16 + mr) * (BST / 2) + 8 * hlf;
      bf0.q[0] = *(const v4u*)(BsU + b0); bf0.q[1] = *(const v4u*)(BsU + b0 + 4);
      bf1.q[0] = *(const v4u*)(BsU + b1); bf1.q[1] = *(const v4u*)(BsU + b1 + 4);
      bf2.q[0] = *(const v4u*)(BsU + b2); bf2.q[1] = *(const v4u*)(BsU + b2 + 4);
      bf3.q[0] = *(const v4u*)(BsU + b3); bf3.q[1] = *(const v4u*)(BsU + b3 + 4);
      af.q[0]  = *(const v4u*)(AsU + aBase);
      af.q[1]  = *(const v4u*)(AsU + aBase + 4);
    }
    acc[0] = __builtin_amdgcn_wmma_f32_16x16x32_bf16(false, af.v, false, bf0.v,
                                                     (short)0, acc[0], false, false);
    acc[1] = __builtin_amdgcn_wmma_f32_16x16x32_bf16(false, af.v, false, bf1.v,
                                                     (short)0, acc[1], false, false);
    acc[2] = __builtin_amdgcn_wmma_f32_16x16x32_bf16(false, af.v, false, bf2.v,
                                                     (short)0, acc[2], false, false);
    acc[3] = __builtin_amdgcn_wmma_f32_16x16x32_bf16(false, af.v, false, bf3.v,
                                                     (short)0, acc[3], false, false);
  }

  float sg = 1.0f, sgc = 0.0f;
  const bool useGate = (gate != nullptr);
  if (useGate) {
    float gv = gate[0];
    sg = 1.0f / (1.0f + expf(-gv));
    sgc = 1.0f - sg;
  }
#pragma unroll
  for (int nt = 0; nt < 4; ++nt) {
#pragma unroll
    for (int vv = 0; vv < 8; ++vv) {
      int r = rowBase + wv * 16 + vv + hlf * 8;
      int cc = colBase + nt * 16 + mr;
      if (r < M) {
        float val = acc[nt][vv];
        if (bias) val += bias[cc];
        size_t off = (size_t)r * ldc + cc;
        if (beta) val += C[off];
        if (useGate) val = sg * val + sgc * skip[off];
        if (out_act == 1) val = fmaxf(val, 0.0f);
        C[off] = val;
      }
    }
  }
}

// ---------------- edge pass 1: logits + per-(dst,head) running max ----------------
// wave per edge: logit[h] = (k[s,h,:] . qt[d,h,:]) * p[h] / sqrt(64)
__global__ void edge_logits(const int* __restrict__ ei,
                            const float* __restrict__ kqv_src,
                            const float* __restrict__ qt,
                            const float* __restrict__ p,
                            float* __restrict__ logits,
                            float* __restrict__ maxlog, int E) {
  int wid = (blockIdx.x * blockDim.x + threadIdx.x) >> 5;
  int lane = threadIdx.x & 31;
  if (wid >= E) return;
  int s = ei[wid];
  int d = ei[E + wid];
  const float* krow = kqv_src + (size_t)s * 768;  // k block at offset 0
  const float* qrow = qt + (size_t)d * 256;
  int f0 = lane * 8;
  float sum = 0.0f;
#pragma unroll
  for (int j = 0; j < 8; ++j) sum += krow[f0 + j] * qrow[f0 + j];
  sum += __shfl_down(sum, 4, 8);
  sum += __shfl_down(sum, 2, 8);
  sum += __shfl_down(sum, 1, 8);
  if ((lane & 7) == 0) {
    int h = lane >> 3;
    float lg = sum * p[h] * 0.125f;  // 1/sqrt(64)
    logits[(size_t)wid * 4 + h] = lg;
    atomicMaxF(&maxlog[(size_t)d * 4 + h], lg);
  }
}

// ---------------- edge pass 2: exp(logit - max), accumulate denominator ----------
__global__ void edge_exp(const int* __restrict__ ei,
                         float* __restrict__ logits,
                         const float* __restrict__ maxlog,
                         float* __restrict__ sumexp, int E) {
  int t = blockIdx.x * blockDim.x + threadIdx.x;
  if (t >= 4 * E) return;
  int e = t >> 2, h = t & 3;
  int d = ei[E + e];
  float ex = expf(logits[t] - maxlog[(size_t)d * 4 + h]);
  logits[t] = ex;
  atomicAdd(&sumexp[(size_t)d * 4 + h], ex);
}

// ---------------- edge pass 3: aggraw[d] += alpha * v[s] ----------------
__global__ void edge_scatter(const int* __restrict__ ei,
                             const float* __restrict__ kqv_src,
                             const float* __restrict__ logits,
                             const float* __restrict__ sumexp,
                             float* __restrict__ aggraw, int E) {
  int wid = (blockIdx.x * blockDim.x + threadIdx.x) >> 5;
  int lane = threadIdx.x & 31;
  if (wid >= E) return;
  int s = ei[wid];
  int d = ei[E + wid];
  int f0 = lane * 8;
  int h = lane >> 3;
  float alpha = logits[(size_t)wid * 4 + h] /
                (sumexp[(size_t)d * 4 + h] + 1e-16f);
  const float* vrow = kqv_src + (size_t)s * 768 + 512;  // v block
  float* arow = aggraw + (size_t)d * 256;
#pragma unroll
  for (int j = 0; j < 8; ++j) atomicAdd(&arow[f0 + j], alpha * vrow[f0 + j]);
}

// ---------------- misc small kernels ----------------
__global__ void fill_f32(float* p, float v, int n) {
  int i = blockIdx.x * blockDim.x + threadIdx.x;
  if (i < n) p[i] = v;
}

__global__ void colmean(const float* __restrict__ Z, float* __restrict__ g, int rows) {
  int r0 = blockIdx.x * 256;
  int t = threadIdx.x;
  float s = 0.0f;
  for (int i = 0; i < 256; ++i) {
    int r = r0 + i;
    if (r < rows) s += Z[(size_t)r * 256 + t];
  }
  atomicAdd(&g[t], s * (1.0f / 50000.0f));
}

__global__ void mlp_head(const float* __restrict__ g,
                         const float* __restrict__ W1, const float* __restrict__ b1,
                         const float* __restrict__ W2, const float* __restrict__ b2,
                         float* __restrict__ out) {
  __shared__ float sg[256];
  __shared__ float sh[256];
  int t = threadIdx.x;
  sg[t] = g[t];
  __syncthreads();
  float a = b1[t];
  for (int j = 0; j < 256; ++j) a += sg[j] * W1[j * 256 + t];
  sh[t] = fmaxf(a, 0.0f) * W2[t];
  __syncthreads();
  for (int off = 128; off > 0; off >>= 1) {
    if (t < off) sh[t] += sh[t + off];
    __syncthreads();
  }
  if (t == 0) out[0] = sh[0] + b2[0];
}

// ---------------- host orchestration ----------------
static inline void launch_gemm(hipStream_t st, const float* A, int lda,
                               const float* B, int ldb, int transB,
                               const float* bias, const float* skip, const float* gate,
                               float* C, int ldc, int M, int N, int K,
                               int inAct, int outAct, int beta) {
  dim3 grid(N / 64, (M + 63) / 64), blk(128);
  gemm_wmma<<<grid, blk, 0, st>>>(A, lda, B, ldb, transB, bias, skip, gate,
                                  C, ldc, M, N, K, inAct, outAct, beta);
}

extern "C" void kernel_launch(void* const* d_in, const int* in_sizes, int n_in,
                              void* d_out, int out_size, void* d_ws, size_t ws_size,
                              hipStream_t stream) {
  (void)in_sizes; (void)n_in; (void)out_size; (void)ws_size;
  auto P = [&](int i) -> const float* { return (const float*)d_in[i]; };
  const float* x_gene = P(0);
  const float* x_snp  = P(1);
  const int* ei_sg = (const int*)d_in[2];
  const int* ei_gg = (const int*)d_in[3];
  const int* ei_gs = (const int*)d_in[4];
  // params flattened in insertion order:
  // 5..8  lin_in {gene{W,b}, snp{W,b}}   9..12 lin_out {gene{W,b}, snp{W,b}}
  // layer L base=13+19L: kqv g{W,b} s{W,b}(+0..3), out g{W,b} s{W,b}(+4..7),
  //   skip g,s(+8,9), rel sg{a,m,p}(+10..12) gg(+13..15) gs(+16..18)
  // 51..54 mlp {W1,b1,W2,b2}

  float* ws = (float*)d_ws;
  size_t o = 0;
  auto alloc = [&](size_t n) { float* p = ws + o; o += n; return p; };
  float* xa_g = alloc((size_t)NGENE * 256);
  float* xb_g = alloc((size_t)NGENE * 256);
  float* xa_s = alloc((size_t)NSNP * 256);
  float* xb_s = alloc((size_t)NSNP * 256);
  float* kqv_g = alloc((size_t)NGENE * 768);
  float* kqv_s = alloc((size_t)NSNP * 768);
  float* qt0 = alloc((size_t)NGENE * 256);  // rel sg (dst=gene); reused as aggraw_sg
  float* qt1 = alloc((size_t)NGENE * 256);  // rel gg (dst=gene); reused as aggraw_gg
  float* qt2 = alloc((size_t)NSNP * 256);   // rel gs (dst=snp);  reused as aggraw_gs
  float* agg_g = alloc((size_t)NGENE * 256);
  float* agg_s = alloc((size_t)NSNP * 256);
  float* lg_sg = alloc((size_t)NE * 4);
  float* lg_gg = alloc((size_t)NE * 4);
  float* lg_gs = alloc((size_t)NE * 4);
  float* ml_g = alloc((size_t)NGENE * 4);
  float* se_g = alloc((size_t)NGENE * 4);
  float* ml_s = alloc((size_t)NSNP * 4);
  float* se_s = alloc((size_t)NSNP * 4);
  float* z    = kqv_g;  // alias: lin_out output after layers done with kqv
  float* gvec = kqv_s;  // alias: 256-float mean vector

  const int EB = NE / 8;  // wave-per-edge blocks of 8 waves

  // ---- input projections (relu) ----
  launch_gemm(stream, x_gene, 128, P(5), 256, 0, P(6), nullptr, nullptr,
              xa_g, 256, NGENE, 256, 128, 0, 1, 0);
  launch_gemm(stream, x_snp, 64, P(7), 256, 0, P(8), nullptr, nullptr,
              xa_s, 256, NSNP, 256, 64, 0, 1, 0);

  for (int L = 0; L < 2; ++L) {
    const int base = 13 + L * 19;
    float* cur_g = L ? xb_g : xa_g;
    float* cur_s = L ? xb_s : xa_s;
    float* nxt_g = L ? xa_g : xb_g;
    float* nxt_s = L ? xa_s : xb_s;
    const float* a_sg = P(base + 10), *m_sg = P(base + 11), *p_sg = P(base + 12);
    const float* a_gg = P(base + 13), *m_gg = P(base + 14), *p_gg = P(base + 15);
    const float* a_gs = P(base + 16), *m_gs = P(base + 17), *p_gs = P(base + 18);

    // kqv = x @ Wkqv + b   [N,768]
    launch_gemm(stream, cur_g, 256, P(base + 0), 768, 0, P(base + 1), nullptr, nullptr,
                kqv_g, 768, NGENE, 768, 256, 0, 0, 0);
    launch_gemm(stream, cur_s, 256, P(base + 2), 768, 0, P(base + 3), nullptr, nullptr,
                kqv_s, 768, NSNP, 768, 256, 0, 0, 0);

    // qt[d,h,:] = a[h] @ q[d,h,:]  -> per-head GEMM  q @ a[h]^T   (bilinear trick)
    for (int h = 0; h < 4; ++h) {
      launch_gemm(stream, kqv_g + 256 + h * 64, 768, a_sg + h * 4096, 64, 1,
                  nullptr, nullptr, nullptr, qt0 + h * 64, 256, NGENE, 64, 64, 0, 0, 0);
      launch_gemm(stream, kqv_g + 256 + h * 64, 768, a_gg + h * 4096, 64, 1,
                  nullptr, nullptr, nullptr, qt1 + h * 64, 256, NGENE, 64, 64, 0, 0, 0);
      launch_gemm(stream, kqv_s + 256 + h * 64, 768, a_gs + h * 4096, 64, 1,
                  nullptr, nullptr, nullptr, qt2 + h * 64, 256, NSNP, 64, 64, 0, 0, 0);
    }

    // softmax stats init
    fill_f32<<<(NGENE * 4 + 255) / 256, 256, 0, stream>>>(ml_g, -1e30f, NGENE * 4);
    fill_f32<<<(NSNP * 4 + 255) / 256, 256, 0, stream>>>(ml_s, -1e30f, NSNP * 4);
    hipMemsetAsync(se_g, 0, (size_t)NGENE * 4 * sizeof(float), stream);
    hipMemsetAsync(se_s, 0, (size_t)NSNP * 4 * sizeof(float), stream);

    // pass 1: logits + max  (src kqv: sg<-snp, gg<-gene, gs<-gene)
    edge_logits<<<EB, 256, 0, stream>>>(ei_sg, kqv_s, qt0, p_sg, lg_sg, ml_g, NE);
    edge_logits<<<EB, 256, 0, stream>>>(ei_gg, kqv_g, qt1, p_gg, lg_gg, ml_g, NE);
    edge_logits<<<EB, 256, 0, stream>>>(ei_gs, kqv_g, qt2, p_gs, lg_gs, ml_s, NE);

    // pass 2: exp + denominator (joint softmax per dst node across relations)
    edge_exp<<<(4 * NE + 255) / 256, 256, 0, stream>>>(ei_sg, lg_sg, ml_g, se_g, NE);
    edge_exp<<<(4 * NE + 255) / 256, 256, 0, stream>>>(ei_gg, lg_gg, ml_g, se_g, NE);
    edge_exp<<<(4 * NE + 255) / 256, 256, 0, stream>>>(ei_gs, lg_gs, ml_s, se_s, NE);

    // qt buffers become aggraw buffers
    hipMemsetAsync(qt0, 0, (size_t)(2 * NGENE + NSNP) * 256 * sizeof(float), stream);

    // pass 3: scatter alpha * v[src]
    edge_scatter<<<EB, 256, 0, stream>>>(ei_sg, kqv_s, lg_sg, se_g, qt0, NE);
    edge_scatter<<<EB, 256, 0, stream>>>(ei_gg, kqv_g, lg_gg, se_g, qt1, NE);
    edge_scatter<<<EB, 256, 0, stream>>>(ei_gs, kqv_g, lg_gs, se_s, qt2, NE);

    // agg = sum_rel aggraw_rel @ m_rel  (per head, beta accumulates 2nd relation)
    for (int h = 0; h < 4; ++h) {
      launch_gemm(stream, qt0 + h * 64, 256, m_sg + h * 4096, 64, 0,
                  nullptr, nullptr, nullptr, agg_g + h * 64, 256, NGENE, 64, 64, 0, 0, 0);
      launch_gemm(stream, qt1 + h * 64, 256, m_gg + h * 4096, 64, 0,
                  nullptr, nullptr, nullptr, agg_g + h * 64, 256, NGENE, 64, 64, 0, 0, 1);
      launch_gemm(stream, qt2 + h * 64, 256, m_gs + h * 4096, 64, 0,
                  nullptr, nullptr, nullptr, agg_s + h * 64, 256, NSNP, 64, 64, 0, 0, 0);
    }

    // x_next = relu( sig(skip)*(gelu(agg) @ Wout + b) + (1-sig)*x )
    launch_gemm(stream, agg_g, 256, P(base + 4), 256, 0, P(base + 5), cur_g, P(base + 8),
                nxt_g, 256, NGENE, 256, 256, 2, 1, 0);
    launch_gemm(stream, agg_s, 256, P(base + 6), 256, 0, P(base + 7), cur_s, P(base + 9),
                nxt_s, 256, NSNP, 256, 256, 2, 1, 0);
  }

  // head: z = x_gene_final @ Wout_gene + b ; g = mean(z, axis=0) ; mlp
  launch_gemm(stream, xa_g, 256, P(9), 256, 0, P(10), nullptr, nullptr,
              z, 256, NGENE, 256, 256, 0, 0, 0);
  hipMemsetAsync(gvec, 0, 256 * sizeof(float), stream);
  colmean<<<(NGENE + 255) / 256, 256, 0, stream>>>(z, gvec, NGENE);
  mlp_head<<<1, 256, 0, stream>>>(gvec, P(51), P(52), P(53), P(54), (float*)d_out);
}
